// GuideBlock_17351667876058
// MI455X (gfx1250) — compile-verified, matrix operand order checked by reading, not run on hardware
//
#include <hip/hip_runtime.h>
#include <hip/hip_bf16.h>

// ---------------------------------------------------------------------------
// GuideBlock for MI455X (gfx1250): all GEMM-shaped work routed through
// v_wmma_f32_16x16x32_bf16 (wave32, 16x16 C/D tiles, fp32 accumulation).
// Register-blocked over M (MREP accumulators share one B fragment) to cut
// activation HBM traffic 4x. Intermediates bf16 NHWC so each WMMA K-chunk
// is exactly two contiguous b128 loads per lane (matches ISA fragment layout).
// ---------------------------------------------------------------------------

typedef __attribute__((ext_vector_type(16))) __bf16 v16bf;
typedef __attribute__((ext_vector_type(8)))  float  v8f;

#define BB 2
#define CCH 64
#define GCH 64
#define OCH 64
#define HH 256
#define WW 256
#define PP (HH * WW)
#define EPSV 1e-5f

// ---------------- layout/packing kernels ----------------------------------

// NCHW fp32 (input, guidance) -> NHWC bf16 concat [b][p][128]
__global__ __launch_bounds__(256)
void pack_kgn_kernel(const float* __restrict__ inp, const float* __restrict__ gui,
                     __hip_bfloat16* __restrict__ out) {
  long idx = (long)blockIdx.x * blockDim.x + threadIdx.x;
  const long total = (long)BB * 128 * PP;
  if (idx >= total) return;
  int b = (int)(idx / (128L * PP));
  int c = (int)((idx / PP) % 128);
  int p = (int)(idx % PP);
  float v = (c < CCH) ? inp[((long)b * CCH + c) * PP + p]
                      : gui[((long)b * GCH + (c - CCH)) * PP + p];
  out[((long)b * PP + p) * 128 + c] = __float2bfloat16(v);
}

// [co][ci][3][3] fp32 -> [co][r][ci] bf16   (K runs r-major, ci-minor)
__global__ __launch_bounds__(256)
void pack_w3_kernel(const float* __restrict__ w, __hip_bfloat16* __restrict__ a,
                    int cout, int cin) {
  long idx = (long)blockIdx.x * blockDim.x + threadIdx.x;
  long total = (long)cout * cin * 9;
  if (idx >= total) return;
  int co = (int)(idx / ((long)cin * 9));
  int rem = (int)(idx % ((long)cin * 9));
  int ci = rem / 9;
  int r  = rem % 9;
  a[((long)co * 9 + r) * cin + ci] = __float2bfloat16(w[((long)co * cin + ci) * 9 + r]);
}

// ---------------- WMMA implicit-GEMM conv ----------------------------------
// out[b, co, p] = bias[co] + sum_{r,ci} A[co][r][ci] * in_nhwc[b][p+off(r)][ci]
// One wave computes MREP stacked 16(M) x 16(N) tiles; B fragment loaded once
// per K-chunk and reused across MREP WMMAs.
// OUTMODE: 0 = bf16 NHWC, 1 = f32 NHWC, 2 = f32 NCHW
template <int CIN, int COUT, int KR, int MREP, bool RELU, int OUTMODE>
__global__ __launch_bounds__(256)
void conv_wmma_kernel(const __hip_bfloat16* __restrict__ in,   // [B][P][CIN]
                      const __hip_bfloat16* __restrict__ A,    // [(b)][COUT][KR*KR][CIN]
                      const float* __restrict__ bias,          // [COUT] or nullptr
                      void* __restrict__ out,
                      int totalWaves, int perBatchA) {
  const int lane = threadIdx.x & 31;
  const int gwave = blockIdx.x * (blockDim.x >> 5) + (threadIdx.x >> 5);
  if (gwave >= totalWaves) return;              // wave-uniform: EXEC stays all-ones
  constexpr int MT = COUT / 16;
  constexpr int MG = MT / MREP;                 // M groups per pixel tile
  const int mt0 = (gwave % MG) * MREP;
  const int nt  = gwave / MG;
  const long n0 = (long)nt * 16;
  const int b  = (int)(n0 / PP);
  const int p0 = (int)(n0 % PP);

  const int m    = lane & 15;
  const int half = lane >> 4;
  const int p = p0 + m;                          // this lane's pixel (N column)
  const int h = p >> 8;                          // WW == 256
  const int w = p & 255;

  v8f acc[MREP];
#pragma unroll
  for (int mi = 0; mi < MREP; ++mi)
#pragma unroll
    for (int e = 0; e < 8; ++e) acc[mi][e] = 0.0f;

  constexpr int K = KR * KR * CIN;
  const __hip_bfloat16* Ab = A + (perBatchA ? (long)b * COUT * KR * KR * CIN : 0);

#pragma unroll 2
  for (int kb = 0; kb < K; kb += 32) {
    const int r   = kb / CIN;                    // constant within chunk (CIN % 32 == 0)
    const int ci0 = kb % CIN;
    const int dh  = r / KR - KR / 2;
    const int dw  = r % KR - KR / 2;

    // B fragment: NHWC activations, zero-padded; two contiguous 16B loads
    // (K = half*8+{0..7} and 16+half*8+{0..7} per the 16-bit fragment layout)
    const int hh = h + dh, ww2 = w + dw;
    uint4 blo = {0u, 0u, 0u, 0u}, bhi = {0u, 0u, 0u, 0u};
    if (hh >= 0 && hh < HH && ww2 >= 0 && ww2 < WW) {
      const __hip_bfloat16* bp =
          in + ((long)b * PP + (long)hh * WW + ww2) * CIN + ci0;
      blo = *(const uint4*)(bp + half * 8);
      bhi = *(const uint4*)(bp + 16 + half * 8);
    }
    v16bf bv;
    ((uint4*)&bv)[0] = blo; ((uint4*)&bv)[1] = bhi;

    // Prefetch next K-chunk's activation line (global_prefetch_b8)
    if (kb + 32 < K) {
      const int kn  = kb + 32;
      const int rn  = kn / CIN;
      const int cn0 = kn % CIN;
      const int hn = h + (rn / KR - KR / 2), wn = w + (rn % KR - KR / 2);
      if (hn >= 0 && hn < HH && wn >= 0 && wn < WW)
        __builtin_prefetch(in + ((long)b * PP + (long)hn * WW + wn) * CIN + cn0, 0, 1);
    }

    // MREP A fragments (weight tiles; small, L2/WGP$-resident) reuse bv
#pragma unroll
    for (int mi = 0; mi < MREP; ++mi) {
      const int row = (mt0 + mi) * 16 + m;
      const __hip_bfloat16* ap = Ab + ((long)row * (KR * KR) + r) * CIN + ci0;
      uint4 alo = *(const uint4*)(ap + half * 8);
      uint4 ahi = *(const uint4*)(ap + 16 + half * 8);
      v16bf av;
      ((uint4*)&av)[0] = alo; ((uint4*)&av)[1] = ahi;
      acc[mi] = __builtin_amdgcn_wmma_f32_16x16x32_bf16(
          /*neg_a=*/false, av, /*neg_b=*/false, bv,
          /*c_mod=*/(short)0, acc[mi], /*reuse_a=*/false, /*reuse_b=*/false);
    }
  }

  // Epilogue. C/D layout: element e -> (M = half*8 + e, N = lane&15)
#pragma unroll
  for (int mi = 0; mi < MREP; ++mi) {
    const int coB = (mt0 + mi) * 16 + half * 8;
    float v[8];
#pragma unroll
    for (int e = 0; e < 8; ++e) {
      float t = acc[mi][e];
      if (bias) t += bias[coB + e];
      if (RELU) t = fmaxf(t, 0.0f);
      v[e] = t;
    }
    if (OUTMODE == 0) {                          // bf16 NHWC, one 16B store/lane
      uint4 q;
      __hip_bfloat16* tp = (__hip_bfloat16*)&q;
#pragma unroll
      for (int e = 0; e < 8; ++e) tp[e] = __float2bfloat16(v[e]);
      __hip_bfloat16* op = (__hip_bfloat16*)out + ((long)b * PP + p) * COUT + coB;
      *(uint4*)op = q;
    } else if (OUTMODE == 1) {                   // f32 NHWC
      float* op = (float*)out + ((long)b * PP + p) * COUT + coB;
#pragma unroll
      for (int e = 0; e < 8; ++e) op[e] = v[e];
    } else {                                     // f32 NCHW (coalesced over p)
      float* op = (float*)out;
#pragma unroll
      for (int e = 0; e < 8; ++e)
        op[((long)b * COUT + coB + e) * PP + p] = v[e];
    }
  }
}

// ---------------- GAP over H,W of f32 NHWC h -> hmean[b][c] ----------------
__global__ __launch_bounds__(256)
void gap_kernel(const float* __restrict__ h, float* __restrict__ hmean) {
  const int b = blockIdx.x / CCH;
  const int c = blockIdx.x % CCH;
  float s = 0.0f;
  for (int p = threadIdx.x; p < PP; p += 256)
    s += h[((long)b * PP + p) * CCH + c];
  __shared__ float red[256];
  red[threadIdx.x] = s;
  __syncthreads();
  for (int st = 128; st > 0; st >>= 1) {
    if (threadIdx.x < st) red[threadIdx.x] += red[threadIdx.x + st];
    __syncthreads();
  }
  if (threadIdx.x == 0) hmean[b * CCH + c] = red[0] * (1.0f / (float)PP);
}

// ---------------- dw = dw_w2 @ hmean + dw_b2, stored bf16 [b][o][c] --------
__global__ __launch_bounds__(256)
void dw_gemv_kernel(const float* __restrict__ w2, const float* __restrict__ b2,
                    const float* __restrict__ hmean, __hip_bfloat16* __restrict__ dwb) {
  int idx = blockIdx.x * blockDim.x + threadIdx.x;    // over B * (O*C) = 8192
  if (idx >= BB * OCH * CCH) return;
  int b = idx / (OCH * CCH);
  int j = idx % (OCH * CCH);                          // j = o*C + c  (reshape order)
  float s = b2[j];
  const float* wr = w2 + (long)j * CCH;
  const float* hm = hmean + b * CCH;
#pragma unroll 8
  for (int ci = 0; ci < CCH; ++ci) s += wr[ci] * hm[ci];
  dwb[(long)b * OCH * CCH + j] = __float2bfloat16(s); // flat == [b][o][c]
}

// ---------------- y[b][p][c] = sum_r input[b,c,p+off(r)] * cw[b][p][c*9+r] -
__global__ __launch_bounds__(256)
void apply_dynamic_kernel(const float* __restrict__ inp,
                          const __hip_bfloat16* __restrict__ cw,
                          __hip_bfloat16* __restrict__ y) {
  long idx = (long)blockIdx.x * blockDim.x + threadIdx.x;  // (b,c,p), p fastest
  const long total = (long)BB * CCH * PP;
  if (idx >= total) return;
  int b = (int)(idx / ((long)CCH * PP));
  int c = (int)((idx / PP) % CCH);
  int p = (int)(idx % PP);
  int h = p >> 8, w = p & 255;
  const __hip_bfloat16* cwp = cw + ((long)b * PP + p) * (9 * CCH) + c * 9;
  const float* irow = inp + ((long)b * CCH + c) * PP;
  float s = 0.0f;
#pragma unroll
  for (int r = 0; r < 9; ++r) {
    int hh = h + r / 3 - 1, ww2 = w + r % 3 - 1;
    if (hh >= 0 && hh < HH && ww2 >= 0 && ww2 < WW)
      s += irow[hh * WW + ww2] * __bfloat162float(cwp[r]);
  }
  y[((long)b * PP + p) * CCH + c] = __float2bfloat16(s);
}

// ---------------- BatchNorm stats over (b, p) per channel o ---------------
__global__ __launch_bounds__(256)
void bn_stats_kernel(const float* __restrict__ z, float* __restrict__ stats) {
  const int o = blockIdx.x;
  float s = 0.0f, s2 = 0.0f;
  const long N = (long)BB * PP;
  for (long i = threadIdx.x; i < N; i += 256) {
    int b = (int)(i / PP);
    int p = (int)(i % PP);
    float v = z[((long)b * OCH + o) * PP + p];
    s += v;
    s2 += v * v;
  }
  __shared__ float r1[256], r2[256];
  r1[threadIdx.x] = s; r2[threadIdx.x] = s2;
  __syncthreads();
  for (int st = 128; st > 0; st >>= 1) {
    if (threadIdx.x < st) { r1[threadIdx.x] += r1[threadIdx.x + st];
                            r2[threadIdx.x] += r2[threadIdx.x + st]; }
    __syncthreads();
  }
  if (threadIdx.x == 0) {
    float mu = r1[0] / (float)N;
    float var = r2[0] / (float)N - mu * mu;
    stats[o * 2] = mu;
    stats[o * 2 + 1] = var;
  }
}

__global__ __launch_bounds__(256)
void bn_apply_kernel(const float* __restrict__ z, const float* __restrict__ stats,
                     const float* __restrict__ gamma, const float* __restrict__ beta,
                     float* __restrict__ out) {
  long idx = (long)blockIdx.x * blockDim.x + threadIdx.x;
  const long total = (long)BB * OCH * PP;
  if (idx >= total) return;
  int o = (int)((idx / PP) % OCH);
  float mu = stats[o * 2], var = stats[o * 2 + 1];
  float v = (z[idx] - mu) * __frsqrt_rn(var + EPSV) * gamma[o] + beta[o];
  out[idx] = fmaxf(v, 0.0f);
}

// ---------------------------------------------------------------------------

extern "C" void kernel_launch(void* const* d_in, const int* in_sizes, int n_in,
                              void* d_out, int out_size, void* d_ws, size_t ws_size,
                              hipStream_t stream) {
  const float* input    = (const float*)d_in[0];
  const float* guidance = (const float*)d_in[1];
  const float* cw_w1 = (const float*)d_in[2];
  const float* cw_b1 = (const float*)d_in[3];
  const float* cw_w2 = (const float*)d_in[4];
  const float* cw_b2 = (const float*)d_in[5];
  const float* dw_w1 = (const float*)d_in[6];
  const float* dw_b1 = (const float*)d_in[7];
  const float* dw_w2 = (const float*)d_in[8];
  const float* dw_b2 = (const float*)d_in[9];
  const float* bn_g  = (const float*)d_in[10];
  const float* bn_b  = (const float*)d_in[11];

  // workspace carve-up (256B aligned regions)
  size_t off = 0;
  auto carve = [&](size_t bytes) -> void* {
    off = (off + 255) & ~(size_t)255;
    void* p = (char*)d_ws + off;
    off += bytes;
    return p;
  };
  __hip_bfloat16* kgn   = (__hip_bfloat16*)carve((size_t)BB * PP * 128 * 2);      // 33.6 MB
  __hip_bfloat16* a_cw1 = (__hip_bfloat16*)carve((size_t)64 * 9 * 128 * 2);
  __hip_bfloat16* a_dw1 = (__hip_bfloat16*)carve((size_t)64 * 9 * 128 * 2);
  __hip_bfloat16* a_cw2 = (__hip_bfloat16*)carve((size_t)576 * 9 * 64 * 2);
  __hip_bfloat16* act1  = (__hip_bfloat16*)carve((size_t)BB * PP * 64 * 2);       // 16.8 MB
  float*          hbuf  = (float*)carve((size_t)BB * PP * 64 * 4);                // 33.6 MB
  __hip_bfloat16* cwbuf = (__hip_bfloat16*)carve((size_t)BB * PP * 576 * 2);      // 151 MB
  float*          hmean = (float*)carve((size_t)BB * 64 * 4);
  __hip_bfloat16* dwbf  = (__hip_bfloat16*)carve((size_t)BB * 64 * 64 * 2);
  __hip_bfloat16* ybuf  = (__hip_bfloat16*)carve((size_t)BB * PP * 64 * 2);       // 16.8 MB
  float*          zbuf  = (float*)carve((size_t)BB * PP * 64 * 4);                // 33.6 MB
  float*          stats = (float*)carve((size_t)64 * 2 * 4);
  (void)ws_size; (void)n_in; (void)in_sizes; (void)out_size;

  // 1. build concat(input,guidance) in bf16 NHWC
  {
    long n = (long)BB * 128 * PP;
    pack_kgn_kernel<<<(int)((n + 255) / 256), 256, 0, stream>>>(input, guidance, kgn);
  }
  // 2. repack conv weights to [co][r][ci] bf16
  pack_w3_kernel<<<(64 * 128 * 9 + 255) / 256, 256, 0, stream>>>(cw_w1, a_cw1, 64, 128);
  pack_w3_kernel<<<(64 * 128 * 9 + 255) / 256, 256, 0, stream>>>(dw_w1, a_dw1, 64, 128);
  pack_w3_kernel<<<(576 * 64 * 9 + 255) / 256, 256, 0, stream>>>(cw_w2, a_cw2, 576, 64);

  const int NT = BB * PP / 16;  // 8192 pixel tiles

  // 3. conv1 (cw path): 64<-128, 3x3, relu, bf16 NHWC out. MREP=4 covers M=64.
  {
    int waves = (64 / 16 / 4) * NT;
    conv_wmma_kernel<128, 64, 3, 4, true, 0><<<(waves + 7) / 8, 256, 0, stream>>>(
        kgn, a_cw1, cw_b1, act1, waves, 0);
  }
  // 4. conv1 (dw path): 64<-128, 3x3, relu, f32 NHWC out (for precise GAP)
  {
    int waves = (64 / 16 / 4) * NT;
    conv_wmma_kernel<128, 64, 3, 4, true, 1><<<(waves + 7) / 8, 256, 0, stream>>>(
        kgn, a_dw1, dw_b1, hbuf, waves, 0);
  }
  // 5. conv2: 576<-64, 3x3, no relu, bf16 NHWC out (per-pixel 3x3 kernels)
  {
    int waves = (576 / 16 / 4) * NT;   // 9 M-groups per pixel tile
    conv_wmma_kernel<64, 576, 3, 4, false, 0><<<(waves + 7) / 8, 256, 0, stream>>>(
        act1, a_cw2, cw_b2, cwbuf, waves, 0);
  }
  // 6. global average pool
  gap_kernel<<<BB * CCH, 256, 0, stream>>>(hbuf, hmean);
  // 7. per-sample 1x1 kernel bank dw[b][o][c] (bias folded), bf16
  dw_gemv_kernel<<<(BB * OCH * CCH + 255) / 256, 256, 0, stream>>>(dw_w2, dw_b2, hmean, dwbf);
  // 8. spatially-varying depthwise 3x3: y = sum_r patches * cw
  {
    long n = (long)BB * CCH * PP;
    apply_dynamic_kernel<<<(int)((n + 255) / 256), 256, 0, stream>>>(input, cwbuf, ybuf);
  }
  // 9. einsum('bchw,boc->bohw') as per-batch WMMA GEMM, f32 NCHW out
  {
    int waves = (64 / 16 / 4) * NT;
    conv_wmma_kernel<64, 64, 1, 4, false, 2><<<(waves + 7) / 8, 256, 0, stream>>>(
        ybuf, dwbf, nullptr, zbuf, waves, 1);
  }
  // 10-11. BatchNorm (batch stats) + ReLU
  bn_stats_kernel<<<OCH, 256, 0, stream>>>(zbuf, stats);
  {
    long n = (long)BB * OCH * PP;
    bn_apply_kernel<<<(int)((n + 255) / 256), 256, 0, stream>>>(
        zbuf, stats, bn_g, bn_b, (float*)d_out);
  }
}
